// TransformerBlock_46256797778282
// MI455X (gfx1250) — compile-verified, matrix operand order checked by reading, not run on hardware
//
#include <hip/hip_runtime.h>
#include <hip/hip_bf16.h>
#include <math.h>

// Problem constants (match reference)
#define B_   4
#define L_   2048
#define D_   768
#define H_   12
#define DF_  3072
#define DK_  64
#define M_   (B_ * L_)   // 8192 rows

typedef __attribute__((ext_vector_type(16))) __bf16 v16bf;
typedef __attribute__((ext_vector_type(8)))  float  v8f;

static __device__ __forceinline__ __bf16 f2bf(float f) {
  unsigned u = __float_as_uint(f);
  u += 0x7fffu + ((u >> 16) & 1u);            // round-to-nearest-even
  unsigned short h = (unsigned short)(u >> 16);
  return __builtin_bit_cast(__bf16, h);
}

// ---------------------------------------------------------------------------
// CDNA5 async global -> LDS copy (ASYNCcnt-tracked) + waits.
// the LDS address comes from the vdst VGPR; low 32 bits of a generic shared pointer are
// the LDS byte offset.
// ---------------------------------------------------------------------------
static __device__ __forceinline__ void async_b128_to_lds(void* lds_ptr, const void* gptr) {
  unsigned lds_off = (unsigned)(uintptr_t)lds_ptr;
  unsigned long long ga = (unsigned long long)(uintptr_t)gptr;
  asm volatile("global_load_async_to_lds_b128 %0, %1, off"
               :: "v"(lds_off), "v"(ga) : "memory");
}
static __device__ __forceinline__ void wait_async_le2() {
  asm volatile("s_wait_asynccnt 0x2" ::: "memory");
}
static __device__ __forceinline__ void wait_async_le1() {
  asm volatile("s_wait_asynccnt 0x1" ::: "memory");
}
static __device__ __forceinline__ void wait_async_le0() {
  asm volatile("s_wait_asynccnt 0x0" ::: "memory");
}

// ---------------------------------------------------------------------------
// Fragment loaders (CDNA5 16-bit A-matrix 16x32 VGPR layout, wave32):
//   lane 0-15  : row M=lane,    K = e (e<8) / 16+e-8 (e>=8)
//   lane 16-31 : row M=lane-16, K = 8+e   / 24+e-8
// Same pattern serves B fragments when B is stored transposed (N x K row-major)
// ---------------------------------------------------------------------------
static __device__ __forceinline__ v16bf load_frag_row(const __bf16* base, int stride,
                                                      int m0, int k0, int lane) {
  int m  = m0 + (lane & 15);
  int kb = k0 + ((lane >> 4) << 3);
  const __bf16* p = base + (long)m * stride + kb;
  v16bf a;
#pragma unroll
  for (int i = 0; i < 8; ++i) { a[i] = p[i]; a[8 + i] = p[16 + i]; }
  return a;
}

// B fragment gathered from a row-major K x N matrix (column gather, stride = N)
static __device__ __forceinline__ v16bf load_frag_col(const __bf16* base, int stride,
                                                      int k0, int n0, int lane) {
  int n  = n0 + (lane & 15);
  int kb = k0 + ((lane >> 4) << 3);
  v16bf b;
#pragma unroll
  for (int i = 0; i < 8; ++i) {
    b[i]     = base[(long)(kb + i) * stride + n];
    b[8 + i] = base[(long)(kb + 16 + i) * stride + n];
  }
  return b;
}

#define WMMA_BF16(A, Bf, C) \
  __builtin_amdgcn_wmma_f32_16x16x32_bf16(false, (A), false, (Bf), (short)0, (C), false, false)

// ---------------------------------------------------------------------------
// Elementwise conversion / weight transpose
// ---------------------------------------------------------------------------
__global__ void k_f32_to_bf16(const float* __restrict__ in, __bf16* __restrict__ out, int n) {
  int i = blockIdx.x * 256 + threadIdx.x;
  if (i < n) out[i] = f2bf(in[i]);
}

// W is (K x N) row-major fp32 -> WT is (N x K) row-major bf16
__global__ void k_transpose_bf16(const float* __restrict__ w, __bf16* __restrict__ wt,
                                 int K, int N) {
  int i = blockIdx.x * 256 + threadIdx.x;
  if (i < K * N) {
    int k = i / N, n = i % N;
    wt[(long)n * K + k] = f2bf(w[i]);
  }
}

// ---------------------------------------------------------------------------
// GEMM: C[M x N] = act(A[M x K](bf16) @ WT[N x K]^T + bias)
// 256 threads = 8 waves; block tile 256 x 64 (each wave owns 32 rows = 2
// M-tiles so every LDS B-fragment feeds 2 WMMAs). The 64 x 32 B strip for
// the current k-chunk is async-DMA'd into double-buffered LDS; A fragments
// are software-pipelined in registers. Steady-state loop is branch-free,
// final k-chunk is peeled.
// ---------------------------------------------------------------------------
__global__ __launch_bounds__(256) void k_gemm_bf16(
    const __bf16* __restrict__ A, const __bf16* __restrict__ WT,
    const float* __restrict__ bias,
    float* __restrict__ out32, __bf16* __restrict__ outbf,
    int N, int K, int gelu) {
  __shared__ __bf16 sB[2][64 * 32];          // 2 x 4KB double buffer

  const int tid  = threadIdx.x;
  const int lane = tid & 31;
  const int w    = tid >> 5;
  const int mA   = blockIdx.y * 256 + w * 32;   // wave's 2 M-tiles: mA, mA+16
  const int n0   = blockIdx.x * 64;

  // per-lane 16B staging slice of the 64x32 strip
  const int srow = tid >> 2;                 // 0..63   (row of WT strip)
  const int scol = (tid & 3) * 8;            // 0,8,16,24 (halfs)
  const __bf16* wsrc = WT + (long)(n0 + srow) * K + scol;

  // prologue: stage k0 = 0 into buffer 0; prime A fragments
  async_b128_to_lds(&sB[0][srow * 32 + scol], wsrc);
  v16bf a0 = load_frag_row(A, K, mA,      0, lane);
  v16bf a1 = load_frag_row(A, K, mA + 16, 0, lane);

  const v8f vzero = {0.f, 0.f, 0.f, 0.f, 0.f, 0.f, 0.f, 0.f};
  v8f c0[4], c1[4];
#pragma unroll
  for (int j = 0; j < 4; ++j) { c0[j] = vzero; c1[j] = vzero; }

  int p = 0;
  for (int k0 = 0; k0 < K - 32; k0 += 32) {
    __syncthreads();                          // everyone done reading sB[p^1]
    async_b128_to_lds(&sB[p ^ 1][srow * 32 + scol], wsrc + k0 + 32);
    v16bf a0n = load_frag_row(A, K, mA,      k0 + 32, lane);   // overlap w/ compute
    v16bf a1n = load_frag_row(A, K, mA + 16, k0 + 32, lane);
    wait_async_le1();                         // sB[p] complete (in-order)
    __syncthreads();                          // ... for all waves
#pragma unroll
    for (int j = 0; j < 4; ++j) {
      v16bf b = load_frag_row(&sB[p][0], 32, j * 16, 0, lane);
      c0[j] = WMMA_BF16(a0, b, c0[j]);
      c1[j] = WMMA_BF16(a1, b, c1[j]);
    }
    a0 = a0n; a1 = a1n;
    p ^= 1;
  }
  // epilogue: final k-chunk (already staged into sB[p])
  wait_async_le0();
  __syncthreads();
#pragma unroll
  for (int j = 0; j < 4; ++j) {
    v16bf b = load_frag_row(&sB[p][0], 32, j * 16, 0, lane);
    c0[j] = WMMA_BF16(a0, b, c0[j]);
    c1[j] = WMMA_BF16(a1, b, c1[j]);
  }

  // C layout: VGPR r -> row +r + 8*(lane>=16); col n = lane&15
  const int mr = mA + ((lane >> 4) << 3);
  const int nc = n0 + (lane & 15);
#pragma unroll
  for (int t = 0; t < 2; ++t) {
#pragma unroll
    for (int j = 0; j < 4; ++j) {
#pragma unroll
      for (int r = 0; r < 8; ++r) {
        int row = mr + t * 16 + r;
        int col = nc + j * 16;
        float v = (t == 0 ? c0[j][r] : c1[j][r]) + bias[col];
        if (gelu) v = 0.5f * v * (1.0f + erff(v * 0.70710678118654752f));
        if (out32) out32[(long)row * N + col] = v;
        if (outbf) outbf[(long)row * N + col] = f2bf(v);
      }
    }
  }
}

// ---------------------------------------------------------------------------
// Flash attention. 8 waves per block, each owns one 16-row Q tile of one
// (b,h). KV is consumed in 32-row chunks; each chunk's K and V tiles
// (4KB each) are async-DMA'd into double-buffered LDS shared by all waves.
// Final chunk handled in-loop (no prefetch, wait 0); mask loads hoisted to overlap the score WMMAs.
// ---------------------------------------------------------------------------
__global__ __launch_bounds__(256) void k_attn(
    const __bf16* __restrict__ q, const __bf16* __restrict__ kmat,
    const __bf16* __restrict__ v, const int* __restrict__ mask,
    __bf16* __restrict__ o) {
  __shared__ __bf16 sK[2][32 * 64];          // 2 x 4KB
  __shared__ __bf16 sV[2][32 * 64];          // 2 x 4KB
  __shared__ __bf16 pbuf[8][16 * 32];        // per-wave P restage (8KB)

  const int tid  = threadIdx.x;
  const int lane = tid & 31;
  const int w    = tid >> 5;
  const int bh   = blockIdx.y;
  const int b    = bh / H_;
  const int h    = bh % H_;
  const int q0   = (blockIdx.x * 8 + w) * 16;

  const __bf16* qb = q    + (long)b * L_ * D_ + h * DK_;
  const __bf16* kb = kmat + (long)b * L_ * D_ + h * DK_;
  const __bf16* vb = v    + (long)b * L_ * D_ + h * DK_;
  const int*    mb = mask + (long)b * L_ * L_;

  // per-lane 16B staging slice of a 32 x 64 tile
  const int srow = tid >> 3;                 // 0..31
  const int scol = (tid & 7) * 8;            // halfs

  // prologue: stage kv = 0 into buffer 0
  async_b128_to_lds(&sK[0][srow * 64 + scol], kb + (long)srow * D_ + scol);
  async_b128_to_lds(&sV[0][srow * 64 + scol], vb + (long)srow * D_ + scol);

  const v8f vzero = {0.f, 0.f, 0.f, 0.f, 0.f, 0.f, 0.f, 0.f};
  v8f acc[4];
#pragma unroll
  for (int j = 0; j < 4; ++j) acc[j] = vzero;
  float mrow[8], lrow[8];
#pragma unroll
  for (int r = 0; r < 8; ++r) { mrow[r] = -1e30f; lrow[r] = 0.f; }

  // Q fragments stay resident
  const v16bf qf0 = load_frag_row(qb, D_, q0, 0,  lane);
  const v16bf qf1 = load_frag_row(qb, D_, q0, 32, lane);

  const int mloc = (lane >> 4) << 3;         // row base for C-layout lanes
  const int nn   = lane & 15;

  int p = 0;
  for (int kv = 0; kv < L_; kv += 32) {
    const bool more = (kv + 32) < L_;
    __syncthreads();                          // done reading buffers p^1
    if (more) {
      async_b128_to_lds(&sK[p ^ 1][srow * 64 + scol],
                        kb + (long)(kv + 32 + srow) * D_ + scol);
      async_b128_to_lds(&sV[p ^ 1][srow * 64 + scol],
                        vb + (long)(kv + 32 + srow) * D_ + scol);
      wait_async_le2();                       // buffers p complete (in-order)
    } else {
      wait_async_le0();
    }
    __syncthreads();

    // hoist mask loads so they overlap the score WMMAs
    int mv0[8], mv1[8];
#pragma unroll
    for (int r = 0; r < 8; ++r) {
      long qoff = (long)(q0 + mloc + r) * L_ + kv + nn;
      mv0[r] = mb[qoff];
      mv1[r] = mb[qoff + 16];
    }

    // S = Q (16x64) @ K^T : two 16x16 score tiles, 2 WMMAs each over DK
    v8f s0 = vzero, s1 = vzero;
    s0 = WMMA_BF16(qf0, load_frag_row(&sK[p][0], 64, 0,  0,  lane), s0);
    s0 = WMMA_BF16(qf1, load_frag_row(&sK[p][0], 64, 0,  32, lane), s0);
    s1 = WMMA_BF16(qf0, load_frag_row(&sK[p][0], 64, 16, 0,  lane), s1);
    s1 = WMMA_BF16(qf1, load_frag_row(&sK[p][0], 64, 16, 32, lane), s1);

    // online softmax over this 16x32 chunk
#pragma unroll
    for (int r = 0; r < 8; ++r) {
      float x0 = s0[r] * 0.125f;                       // 1/sqrt(64)
      float x1 = s1[r] * 0.125f;
      if (mv0[r] == 0) x0 = -1e30f;
      if (mv1[r] == 0) x1 = -1e30f;

      float tmax = fmaxf(x0, x1);
#pragma unroll
      for (int sh = 1; sh < 16; sh <<= 1) tmax = fmaxf(tmax, __shfl_xor(tmax, sh, 32));
      float mnew  = fmaxf(mrow[r], tmax);
      float alpha = __expf(mrow[r] - mnew);
      float p0 = __expf(x0 - mnew);
      float p1 = __expf(x1 - mnew);
      float rs = p0 + p1;
#pragma unroll
      for (int sh = 1; sh < 16; sh <<= 1) rs += __shfl_xor(rs, sh, 32);
      lrow[r] = lrow[r] * alpha + rs;
      mrow[r] = mnew;
#pragma unroll
      for (int j = 0; j < 4; ++j) acc[j][r] *= alpha;

      pbuf[w][(mloc + r) * 32 + nn]      = f2bf(p0);
      pbuf[w][(mloc + r) * 32 + 16 + nn] = f2bf(p1);
    }

    // P (16x32) @ V (32x64): restage P via LDS into A-fragment layout
    v16bf pf = load_frag_row(&pbuf[w][0], 32, 0, 0, lane);
#pragma unroll
    for (int j = 0; j < 4; ++j) {
      v16bf vf = load_frag_col(&sV[p][0], 64, 0, j * 16, lane);
      acc[j] = WMMA_BF16(pf, vf, acc[j]);
    }
    p ^= 1;
  }

  // normalize + store bf16
#pragma unroll
  for (int j = 0; j < 4; ++j) {
#pragma unroll
    for (int r = 0; r < 8; ++r) {
      long row = (long)b * L_ + q0 + mloc + r;
      o[row * D_ + h * DK_ + j * 16 + nn] = f2bf(acc[j][r] / lrow[r]);
    }
  }
}

// ---------------------------------------------------------------------------
// y = LN(a + b): mean, std with ddof=1, eps added to std (reference-faithful).
// 256 threads per 768-element row.
// ---------------------------------------------------------------------------
__global__ __launch_bounds__(256) void k_add_ln(
    const float* __restrict__ a, const float* __restrict__ bres,
    const float* __restrict__ g, const float* __restrict__ beta,
    float* __restrict__ out32, __bf16* __restrict__ outbf) {
  __shared__ float red[256];
  const int row = blockIdx.x;
  const int t   = threadIdx.x;
  const float* pa = a    + (long)row * D_;
  const float* pb = bres + (long)row * D_;

  float vbuf[3];
  float s = 0.f, s2 = 0.f;
#pragma unroll
  for (int i = 0; i < 3; ++i) {
    int c = t + i * 256;
    float vv = pa[c] + pb[c];
    vbuf[i] = vv; s += vv; s2 += vv * vv;
  }
  red[t] = s; __syncthreads();
  for (int off = 128; off > 0; off >>= 1) { if (t < off) red[t] += red[t + off]; __syncthreads(); }
  float sum = red[0]; __syncthreads();
  red[t] = s2; __syncthreads();
  for (int off = 128; off > 0; off >>= 1) { if (t < off) red[t] += red[t + off]; __syncthreads(); }
  float sumsq = red[0];

  float mean = sum * (1.0f / D_);
  float var  = fmaxf((sumsq - (float)D_ * mean * mean) / (float)(D_ - 1), 0.f);
  float inv  = 1.0f / (sqrtf(var) + 1e-6f);
#pragma unroll
  for (int i = 0; i < 3; ++i) {
    int c = t + i * 256;
    float y = g[c] * (vbuf[i] - mean) * inv + beta[c];
    out32[(long)row * D_ + c] = y;
    if (outbf) outbf[(long)row * D_ + c] = f2bf(y);
  }
}

// ---------------------------------------------------------------------------
// Host orchestration
// ---------------------------------------------------------------------------
extern "C" void kernel_launch(void* const* d_in, const int* in_sizes, int n_in,
                              void* d_out, int out_size, void* d_ws, size_t ws_size,
                              hipStream_t stream) {
  const float* x    = (const float*)d_in[0];
  const int*   mask = (const int*)  d_in[1];
  const float* Wq = (const float*)d_in[2];  const float* bq = (const float*)d_in[3];
  const float* Wk = (const float*)d_in[4];  const float* bk = (const float*)d_in[5];
  const float* Wv = (const float*)d_in[6];  const float* bv = (const float*)d_in[7];
  const float* Wo = (const float*)d_in[8];  const float* bo = (const float*)d_in[9];
  const float* W1 = (const float*)d_in[10]; const float* b1 = (const float*)d_in[11];
  const float* W2 = (const float*)d_in[12]; const float* b2 = (const float*)d_in[13];
  const float* g1 = (const float*)d_in[14]; const float* be1 = (const float*)d_in[15];
  const float* g2 = (const float*)d_in[16]; const float* be2 = (const float*)d_in[17];

  char* ws = (char*)d_ws;
  size_t off = 0;
  auto alloc = [&](size_t bytes) -> char* {
    char* p = ws + off;
    off = (off + bytes + 255) & ~(size_t)255;
    return p;
  };
  __bf16* xb  = (__bf16*)alloc((size_t)M_ * D_ * 2);
  __bf16* qb  = (__bf16*)alloc((size_t)M_ * D_ * 2);
  __bf16* kb  = (__bf16*)alloc((size_t)M_ * D_ * 2);
  __bf16* vb  = (__bf16*)alloc((size_t)M_ * D_ * 2);
  __bf16* ob  = (__bf16*)alloc((size_t)M_ * D_ * 2);
  __bf16* x1b = (__bf16*)alloc((size_t)M_ * D_ * 2);
  __bf16* h1b = (__bf16*)alloc((size_t)M_ * DF_ * 2);
  __bf16* WqT = (__bf16*)alloc((size_t)D_ * D_ * 2);
  __bf16* WkT = (__bf16*)alloc((size_t)D_ * D_ * 2);
  __bf16* WvT = (__bf16*)alloc((size_t)D_ * D_ * 2);
  __bf16* WoT = (__bf16*)alloc((size_t)D_ * D_ * 2);
  __bf16* W1T = (__bf16*)alloc((size_t)D_ * DF_ * 2);
  __bf16* W2T = (__bf16*)alloc((size_t)DF_ * D_ * 2);
  float*  sa32 = (float*)alloc((size_t)M_ * D_ * 4);
  float*  x132 = (float*)alloc((size_t)M_ * D_ * 4);
  float*  ff32 = (float*)alloc((size_t)M_ * D_ * 4);

  // 1) x -> bf16
  {
    int n = M_ * D_;
    k_f32_to_bf16<<<(n + 255) / 256, 256, 0, stream>>>(x, xb, n);
  }
  // 2) weight transposes to (N x K) bf16
  {
    int n = D_ * D_;
    k_transpose_bf16<<<(n + 255) / 256, 256, 0, stream>>>(Wq, WqT, D_, D_);
    k_transpose_bf16<<<(n + 255) / 256, 256, 0, stream>>>(Wk, WkT, D_, D_);
    k_transpose_bf16<<<(n + 255) / 256, 256, 0, stream>>>(Wv, WvT, D_, D_);
    k_transpose_bf16<<<(n + 255) / 256, 256, 0, stream>>>(Wo, WoT, D_, D_);
    int n2 = D_ * DF_;
    k_transpose_bf16<<<(n2 + 255) / 256, 256, 0, stream>>>(W1, W1T, D_, DF_);
    k_transpose_bf16<<<(n2 + 255) / 256, 256, 0, stream>>>(W2, W2T, DF_, D_);
  }
  // 3) Q, K, V projections (bf16 outputs)
  {
    dim3 grid(D_ / 64, M_ / 256), blk(256);
    k_gemm_bf16<<<grid, blk, 0, stream>>>(xb, WqT, bq, nullptr, qb, D_, D_, 0);
    k_gemm_bf16<<<grid, blk, 0, stream>>>(xb, WkT, bk, nullptr, kb, D_, D_, 0);
    k_gemm_bf16<<<grid, blk, 0, stream>>>(xb, WvT, bv, nullptr, vb, D_, D_, 0);
  }
  // 4) attention
  {
    dim3 grid(L_ / 128, B_ * H_), blk(256);
    k_attn<<<grid, blk, 0, stream>>>(qb, kb, vb, mask, ob);
  }
  // 5) output projection (fp32 out)
  {
    dim3 grid(D_ / 64, M_ / 256), blk(256);
    k_gemm_bf16<<<grid, blk, 0, stream>>>(ob, WoT, bo, sa32, nullptr, D_, D_, 0);
  }
  // 6) x1 = LN(x + sa)
  k_add_ln<<<M_, 256, 0, stream>>>(x, sa32, g1, be1, x132, x1b);
  // 7) FFN up + GELU (bf16 out)
  {
    dim3 grid(DF_ / 64, M_ / 256), blk(256);
    k_gemm_bf16<<<grid, blk, 0, stream>>>(x1b, W1T, b1, nullptr, h1b, DF_, D_, 1);
  }
  // 8) FFN down (fp32 out)
  {
    dim3 grid(D_ / 64, M_ / 256), blk(256);
    k_gemm_bf16<<<grid, blk, 0, stream>>>(h1b, W2T, b2, ff32, nullptr, D_, DF_, 0);
  }
  // 9) out = LN(x1 + ff)
  k_add_ln<<<M_, 256, 0, stream>>>(x132, ff32, g2, be2, (float*)d_out, nullptr);
}
